// LinearAttention_75883482186016
// MI455X (gfx1250) — compile-verified, matrix operand order checked by reading, not run on hardware
//
#include <hip/hip_runtime.h>
#include <hip/hip_bf16.h>

// Problem constants from the reference
#define BB 8
#define TT 4096
#define SS 4096
#define FF 64
#define MM 64          // n_omega; 2M = 128 feature dim

typedef __attribute__((ext_vector_type(16))) _Float16 v16h;
typedef __attribute__((ext_vector_type(8)))  float    v8f;

// --------------------------------------------------------------------------
// Kernel 1: phi feature map.
// grid = B*L blocks of 64 threads; block p handles position p, thread m
// handles omega row m. Output (p, 0:128) f16: [exp(+xw), exp(-xw)] * h / 8.
// --------------------------------------------------------------------------
__global__ __launch_bounds__(64)
void phi_kernel(const float* __restrict__ x,      // (B*L, 64)
                const float* __restrict__ omega,  // (64, 64)
                _Float16* __restrict__ out)       // (B*L, 128) f16
{
    __shared__ float xs[FF];
    const int p = blockIdx.x;
    const int m = threadIdx.x;          // 0..63

    xs[m] = x[(size_t)p * FF + m];
    __syncthreads();

    float ss = 0.0f, dot = 0.0f;
    const float* om = omega + (size_t)m * FF;
#pragma unroll
    for (int f = 0; f < FF; ++f) {
        float xv = xs[f];
        ss  += xv * xv;                 // redundant per-thread, but tiny
        dot += xv * om[f];
    }
    // h / sqrt(n_omega) = exp(-ss/2) / sqrt(2) / 8
    const float hs = __expf(-0.5f * ss) * 0.08838834764831845f;
    // Regular (RT) stores: these are re-read immediately by the GEMM and
    // should stay hot in L2 (16 MB working set << 192 MB L2).
    out[(size_t)p * (2 * MM) + m]      = (_Float16)(hs * __expf(dot));
    out[(size_t)p * (2 * MM) + MM + m] = (_Float16)(hs * __expf(-dot));
}

// --------------------------------------------------------------------------
// WMMA fragment load from LDS.
// 16-bit A (and symmetric B) layout for v_wmma_f32_16x16x32_f16:
//   lanes 0-15  : row = lane,    VGPR0-3 hold K=0..7,  VGPR4-7 hold K=16..23
//   lanes 16-31 : row = lane-16, VGPR0-3 hold K=8..15, VGPR4-7 hold K=24..31
// => per lane: two contiguous 8-half (16 B) chunks at kbase(+8) and +16.
// --------------------------------------------------------------------------
#define LDP 136   // padded row length (halves): stride = 68 dwords -> rows hit
                  // distinct 4-bank groups across lanes 0..15 for b128 loads

__device__ __forceinline__ v16h load_frag(const _Float16* __restrict__ smem,
                                          int row0, int kbase)
{
    const int lane = threadIdx.x & 31;
    const int r    = row0 + (lane & 15);
    const int k0   = kbase + ((lane >> 4) << 3);     // +8 for upper half-wave
    const _Float16* p = smem + r * LDP + k0;
    union { v16h v; uint4 u[2]; } t;
    t.u[0] = *(const uint4*)(p);
    t.u[1] = *(const uint4*)(p + 16);
    return t.v;
}

__device__ __forceinline__ v8f wmma_f16(v16h a, v16h b, v8f c)
{
    return __builtin_amdgcn_wmma_f32_16x16x32_f16(
        /*neg_a=*/false, a, /*neg_b=*/false, b,
        /*c_mod=*/(short)0, c, /*reuse_a=*/false, /*reuse_b=*/false);
}

// --------------------------------------------------------------------------
// Kernel 2: batched GEMM  out[b,t,s] = sum_k qphi[b,t,k] * kphi[b,s,k], K=128.
// Workgroup: 256 threads = 8 wave32. Tile: 128 (T) x 64 (S), full K in LDS.
// Wave w computes rows [ (w>>1)*32 , +32 ) x cols [ (w&1)*32 , +32 ):
// 2x2 tiles of 16x16, 4 K-steps -> 16 v_wmma per wave.
// Output stores are non-temporal: 536 MB fp32 stream must not evict the
// 16 MB phi working set from L2 (which feeds every tile re-read).
// --------------------------------------------------------------------------
__global__ __launch_bounds__(256)
void gemm_kernel(const _Float16* __restrict__ qphi,  // (B, T, 128)
                 const _Float16* __restrict__ kphi,  // (B, S, 128)
                 float* __restrict__ out)            // (B, T, S)
{
    __shared__ _Float16 sA[128 * LDP];   // 34816 B
    __shared__ _Float16 sB[ 64 * LDP];   // 17408 B   (total 52224 B < 64 KB)

    const int b     = blockIdx.z;
    const int tBase = blockIdx.y * 128;
    const int sBase = blockIdx.x * 64;
    const int tid   = threadIdx.x;

    // ---- Global -> LDS (coalesced b128 loads). K=128 halves = 16 uint4/row.
    const uint4* gq = (const uint4*)(qphi + ((size_t)b * TT + tBase) * (2 * MM));
#pragma unroll
    for (int i = 0; i < 8; ++i) {                 // 128 rows * 16 chunks / 256
        int idx = tid + i * 256;
        int row = idx >> 4, col = idx & 15;
        *(uint4*)&sA[row * LDP + col * 8] = gq[idx];
    }
    const uint4* gk = (const uint4*)(kphi + ((size_t)b * SS + sBase) * (2 * MM));
#pragma unroll
    for (int i = 0; i < 4; ++i) {                 // 64 rows * 16 chunks / 256
        int idx = tid + i * 256;
        int row = idx >> 4, col = idx & 15;
        *(uint4*)&sB[row * LDP + col * 8] = gk[idx];
    }
    __syncthreads();

    const int wave = tid >> 5;
    const int lane = tid & 31;
    const int wRow = (wave >> 1) << 5;   // 0,32,64,96
    const int wCol = (wave &  1) << 5;   // 0,32

    v8f acc[2][2] = {};

#pragma unroll
    for (int kk = 0; kk < 4; ++kk) {
        const int kb = kk * 32;
        v16h a0 = load_frag(sA, wRow,      kb);
        v16h a1 = load_frag(sA, wRow + 16, kb);
        v16h b0 = load_frag(sB, wCol,      kb);
        v16h b1 = load_frag(sB, wCol + 16, kb);
        acc[0][0] = wmma_f16(a0, b0, acc[0][0]);
        acc[0][1] = wmma_f16(a0, b1, acc[0][1]);
        acc[1][0] = wmma_f16(a1, b0, acc[1][0]);
        acc[1][1] = wmma_f16(a1, b1, acc[1][1]);
    }

    // ---- Store accumulators (non-temporal streaming writes).
    // C/D layout: VGPR r, lanes 0-15 -> row r, col lane; lanes 16-31 -> row 8+r.
    const int colIn  = lane & 15;
    const int rowOff = (lane >> 4) << 3;
    const size_t outBase = ((size_t)b * TT + tBase) * (size_t)SS + sBase;
#pragma unroll
    for (int mi = 0; mi < 2; ++mi) {
#pragma unroll
        for (int ni = 0; ni < 2; ++ni) {
            const int r0 = wRow + mi * 16 + rowOff;
            const int c  = wCol + ni * 16 + colIn;
#pragma unroll
            for (int r = 0; r < 8; ++r) {
                __builtin_nontemporal_store(
                    acc[mi][ni][r],
                    &out[outBase + (size_t)(r0 + r) * SS + c]);
            }
        }
    }
}

// --------------------------------------------------------------------------
extern "C" void kernel_launch(void* const* d_in, const int* in_sizes, int n_in,
                              void* d_out, int out_size, void* d_ws, size_t ws_size,
                              hipStream_t stream)
{
    (void)in_sizes; (void)n_in; (void)out_size; (void)ws_size;

    const float* query = (const float*)d_in[0];   // (B, T, F)
    const float* value = (const float*)d_in[1];   // unused by reference output
    const float* key   = (const float*)d_in[2];   // (B, S, F)
    const float* omega = (const float*)d_in[3];   // (M, F)
    (void)value;

    float* out = (float*)d_out;

    // Workspace: qphi (8 MB f16) + kphi (8 MB f16)
    _Float16* qphi = (_Float16*)d_ws;
    _Float16* kphi = qphi + (size_t)BB * TT * (2 * MM);

    phi_kernel<<<BB * TT, 64, 0, stream>>>(query, omega, qphi);
    phi_kernel<<<BB * SS, 64, 0, stream>>>(key,   omega, kphi);

    dim3 grid(SS / 64, TT / 128, BB);
    gemm_kernel<<<grid, 256, 0, stream>>>(qphi, kphi, out);
}
